// MultiScaleBlock_91053306675739
// MI455X (gfx1250) — compile-verified, matrix operand order checked by reading, not run on hardware
//
#include <hip/hip_runtime.h>
#include <hip/hip_bf16.h>

// ---------------------------------------------------------------------------
// MI455X (gfx1250): bf16 WMMA for all GEMM-class work, fp32 elsewhere.
// Round 2: 128x128 GEMM tiles (8 WMMA/wave/k-step) + double-buffered LDS
// staging via GLOBAL_LOAD_ASYNC_TO_LDS_B128 (ASYNCcnt-pipelined).
// ---------------------------------------------------------------------------

#define DIM   768
#define MID   4
#define NH    12
#define HD    64
#define HID   3072
#define TOK   16384        // 2*8*32*32
#define NWIN  32           // 2 * (8/8)*(32/8)*(32/8)
#define WTOK  512          // 8*8*8 tokens per window

typedef __attribute__((ext_vector_type(16))) __bf16 v16bf;
typedef __attribute__((ext_vector_type(8)))  float  v8f;

union Frag16 { v16bf v; uint4 u[2]; };

static __device__ __forceinline__ v8f v8f_zero() {
    v8f z = {0.f,0.f,0.f,0.f,0.f,0.f,0.f,0.f};
    return z;
}

static __device__ __forceinline__ v8f wmma_bf16(const Frag16& a, const Frag16& b, v8f c) {
    return __builtin_amdgcn_wmma_f32_16x16x32_bf16(false, a.v, false, b.v, (short)0, c,
                                                   false, false);
}

// Async global->LDS 128-bit copy (tracked by ASYNCcnt, no VGPR round trip).
// vdst = LDS byte address (low 32 bits of generic pointer), vaddr = global addr.
static __device__ __forceinline__ void async_ld_b128(const void* gaddr, void* lds) {
    asm volatile("global_load_async_to_lds_b128 %0, %1, off"
                 :: "v"((unsigned)(unsigned long long)lds), "v"(gaddr)
                 : "memory");
}

// window token (win, n) -> flat global token index in (B=2, D=8, H=32, W=32)
static __device__ __forceinline__ int win_tok(int win, int n) {
    int b  = win >> 4;          // 16 windows per batch
    int rm = win & 15;
    int wh = rm >> 2, ww = rm & 3;
    int dz = n >> 6;
    int hy = (n >> 3) & 7;
    int wx = n & 7;
    int d = dz, h = wh * 8 + hy, w = ww * 8 + wx;
    return ((b * 8 + d) * 32 + h) * 32 + w;
}

// ---------------------------------------------------------------------------
// Weight convert: fp32 (K x N) row-major  ->  bf16 (N x K) row-major
// ---------------------------------------------------------------------------
__global__ __launch_bounds__(256) void convert_wT_kernel(const float* __restrict__ w,
                                                         __bf16* __restrict__ wt,
                                                         int K, int N) {
    int idx = blockIdx.x * 256 + threadIdx.x;
    if (idx >= K * N) return;
    int n = idx / K, k = idx % K;
    wt[idx] = (__bf16)w[(size_t)k * N + n];
}

// ---------------------------------------------------------------------------
// block reduction helper (256 threads)
// ---------------------------------------------------------------------------
static __device__ __forceinline__ float block_sum(float v, float* red) {
    int tid = threadIdx.x;
    red[tid] = v;
    __syncthreads();
    for (int s = 128; s > 0; s >>= 1) {
        if (tid < s) red[tid] += red[tid + s];
        __syncthreads();
    }
    float r = red[0];
    __syncthreads();
    return r;
}

// ---------------------------------------------------------------------------
// LayerNorm per token (block = 1 token, 256 threads, 3 elems/thread).
// Optionally fuses the adapter down-projection a1 = relu(h @ aw1 + ab1).
// ---------------------------------------------------------------------------
template<bool ADAPTER>
__global__ __launch_bounds__(256) void ln_kernel(const float* __restrict__ x,
                                                 const float* __restrict__ g,
                                                 const float* __restrict__ b,
                                                 __bf16* __restrict__ hout,
                                                 const float* __restrict__ aw1,
                                                 const float* __restrict__ ab1,
                                                 float* __restrict__ a1out) {
    __shared__ float red[256];
    int t = blockIdx.x, tid = threadIdx.x;
    const float* xr = x + (size_t)t * DIM;
    int c0 = tid, c1 = tid + 256, c2 = tid + 512;
    float v0 = xr[c0], v1 = xr[c1], v2 = xr[c2];
    float s  = block_sum(v0 + v1 + v2, red);
    float ss = block_sum(v0 * v0 + v1 * v1 + v2 * v2, red);
    float mean = s * (1.0f / DIM);
    float var  = ss * (1.0f / DIM) - mean * mean;
    float rstd = rsqrtf(var + 1e-6f);
    float h0 = (v0 - mean) * rstd * g[c0] + b[c0];
    float h1 = (v1 - mean) * rstd * g[c1] + b[c1];
    float h2 = (v2 - mean) * rstd * g[c2] + b[c2];
    __bf16* hr = hout + (size_t)t * DIM;
    hr[c0] = (__bf16)h0; hr[c1] = (__bf16)h1; hr[c2] = (__bf16)h2;
    if (ADAPTER) {
        for (int j = 0; j < MID; ++j) {
            float p = h0 * aw1[c0 * MID + j] + h1 * aw1[c1 * MID + j] + h2 * aw1[c2 * MID + j];
            float tot = block_sum(p, red);
            if (tid == 0) a1out[t * MID + j] = fmaxf(tot + ab1[j], 0.0f);
        }
    }
}

// ---------------------------------------------------------------------------
// Depthwise 3x3x3 conv (SAME) on (2,8,32,32,4) + bias + relu
// ---------------------------------------------------------------------------
__global__ __launch_bounds__(256) void dwconv_kernel(const float* __restrict__ a1,
                                                     const float* __restrict__ acw,
                                                     const float* __restrict__ acb,
                                                     float* __restrict__ a2) {
    int idx = blockIdx.x * 256 + threadIdx.x;
    if (idx >= TOK) return;
    int w  = idx & 31;
    int h  = (idx >> 5) & 31;
    int d  = (idx >> 10) & 7;
    int bz = idx >> 13;
    float acc[MID] = {0.f, 0.f, 0.f, 0.f};
    for (int kd = -1; kd <= 1; ++kd) {
        int dd = d + kd; if (dd < 0 || dd >= 8) continue;
        for (int kh = -1; kh <= 1; ++kh) {
            int hh = h + kh; if (hh < 0 || hh >= 32) continue;
            for (int kw = -1; kw <= 1; ++kw) {
                int wwx = w + kw; if (wwx < 0 || wwx >= 32) continue;
                int src = ((bz * 8 + dd) * 32 + hh) * 32 + wwx;
                int wi  = (((kd + 1) * 3 + (kh + 1)) * 3 + (kw + 1)) * MID;
                for (int ch = 0; ch < MID; ++ch)
                    acc[ch] += a1[src * MID + ch] * acw[wi + ch];
            }
        }
    }
    for (int ch = 0; ch < MID; ++ch)
        a2[idx * MID + ch] = fmaxf(acc[ch] + acb[ch], 0.0f);
}

// ---------------------------------------------------------------------------
// hh = 2*h + relu(a2 @ aw2 + ab2), in place over the bf16 h buffer
// ---------------------------------------------------------------------------
__global__ __launch_bounds__(256) void hh_kernel(__bf16* __restrict__ h,
                                                 const float* __restrict__ a2,
                                                 const float* __restrict__ aw2,
                                                 const float* __restrict__ ab2) {
    int t = blockIdx.x;
    float a0 = a2[t * MID + 0], a1v = a2[t * MID + 1];
    float a2v = a2[t * MID + 2], a3 = a2[t * MID + 3];
    __bf16* hr = h + (size_t)t * DIM;
    for (int c = threadIdx.x; c < DIM; c += 256) {
        float s = a0 * aw2[c] + a1v * aw2[DIM + c] + a2v * aw2[2 * DIM + c]
                + a3 * aw2[3 * DIM + c] + ab2[c];
        s = fmaxf(s, 0.0f);
        float hv = (float)hr[c];
        hr[c] = (__bf16)(2.0f * hv + s);
    }
}

// ---------------------------------------------------------------------------
// Tiled bf16 WMMA GEMM: C[M,N] = act(A[M,K] * B^T[N,K] + bias) (+ residual)
// Block tile 128x128, K-step 32. 8 waves in a 4(M) x 2(N) grid; each wave
// owns a 32x64 slab = 8 WMMA accumulators. LDS double-buffered and filled by
// GLOBAL_LOAD_ASYNC_TO_LDS_B128; the next tile is prefetched while computing
// (async ops complete in order, so s_wait_asynccnt 4 = current tile landed).
// ---------------------------------------------------------------------------
template<int ACT, bool RES, bool OUTBF>
__global__ __launch_bounds__(256) void gemm_bf16_kernel(const __bf16* __restrict__ A,
                                                        const __bf16* __restrict__ B,
                                                        const float* __restrict__ bias,
                                                        const float* __restrict__ res,
                                                        void* __restrict__ outp,
                                                        int M, int N, int K) {
    __shared__ __align__(16) __bf16 As[2][128][32];   // 8KB per buffer
    __shared__ __align__(16) __bf16 Bs[2][128][32];
    int tid  = threadIdx.x;
    int wave = tid >> 5, lane = tid & 31;
    int half = lane >> 4, r = lane & 15, ko = half * 8;
    int wm = wave & 3, wn = wave >> 2;                // 4 x 2 wave grid
    int rowBase = blockIdx.y * 128, colBase = blockIdx.x * 128;

    v8f acc[2][4];
    for (int i = 0; i < 2; ++i)
        for (int j = 0; j < 4; ++j) acc[i][j] = v8f_zero();

    // staging map: 256 threads x (2 rows A + 2 rows B) of 16B each per tile
    int sr = tid >> 2;                 // 0..63
    int sc = (tid & 3) * 8;            // 0,8,16,24
    const __bf16* aP = A + (size_t)rowBase * K;
    const __bf16* bP = B + (size_t)colBase * K;

    auto stage = [&](int buf, int k0) {
        async_ld_b128(aP + (size_t)sr * K + k0 + sc,        &As[buf][sr][sc]);
        async_ld_b128(aP + (size_t)(sr + 64) * K + k0 + sc, &As[buf][sr + 64][sc]);
        async_ld_b128(bP + (size_t)sr * K + k0 + sc,        &Bs[buf][sr][sc]);
        async_ld_b128(bP + (size_t)(sr + 64) * K + k0 + sc, &Bs[buf][sr + 64][sc]);
    };

    stage(0, 0);
    int cur = 0;
    for (int k0 = 0; k0 < K; k0 += 32) {
        if (k0 + 32 < K) {
            stage(cur ^ 1, k0 + 32);                       // prefetch next tile
            asm volatile("s_wait_asynccnt 0x4" ::: "memory");  // current tile done
        } else {
            asm volatile("s_wait_asynccnt 0x0" ::: "memory");
        }
        __syncthreads();                                   // visible to all waves

        Frag16 fa[2], fb[4];
        for (int i = 0; i < 2; ++i) {
            fa[i].u[0] = *(const uint4*)&As[cur][wm * 32 + i * 16 + r][ko];
            fa[i].u[1] = *(const uint4*)&As[cur][wm * 32 + i * 16 + r][16 + ko];
        }
        for (int j = 0; j < 4; ++j) {
            fb[j].u[0] = *(const uint4*)&Bs[cur][wn * 64 + j * 16 + r][ko];
            fb[j].u[1] = *(const uint4*)&Bs[cur][wn * 64 + j * 16 + r][16 + ko];
        }
        for (int i = 0; i < 2; ++i)
            for (int j = 0; j < 4; ++j)
                acc[i][j] = wmma_bf16(fa[i], fb[j], acc[i][j]);

        __syncthreads();               // all waves done reading before next stage
        cur ^= 1;
    }

    for (int i = 0; i < 2; ++i) {
        int outRow0 = rowBase + wm * 32 + i * 16;
        for (int e = 0; e < 8; ++e) {
            int row = outRow0 + e + 8 * half;
            for (int j = 0; j < 4; ++j) {
                int col = colBase + wn * 64 + j * 16 + r;
                float v = acc[i][j][e] + bias[col];
                if (ACT == 1) v = 0.5f * v * (1.0f + erff(v * 0.70710678f));  // exact gelu
                if (RES) v += res[(size_t)row * N + col];
                size_t o = (size_t)row * N + col;
                if (OUTBF) ((__bf16*)outp)[o] = (__bf16)v;
                else       ((float*)outp)[o]  = v;
            }
        }
    }
}

// ---------------------------------------------------------------------------
// Windowed flash attention. Block = (window*NH + head, qChunk of 128 rows).
// 8 waves; each wave owns 16 query rows. K-chunk 64 keys staged in LDS,
// V staged transposed. Online softmax; P re-enters WMMA A-layout via
// wave-private LDS guarded by s_wait_dscnt.
// ---------------------------------------------------------------------------
__global__ __launch_bounds__(256) void attn_kernel(const __bf16* __restrict__ qkv,
                                                   __bf16* __restrict__ out) {
    __shared__ __align__(16) __bf16 Ksh[64][64];      // [key][hd]
    __shared__ __align__(16) __bf16 Vt[64][64];       // [hd][key]
    __shared__ __align__(16) __bf16 Ps[8][16][64];    // per-wave P staging

    int win  = blockIdx.x / NH;
    int head = blockIdx.x % NH;
    int qb   = blockIdx.y;                            // 4 chunks of 128 q rows
    int tid  = threadIdx.x;
    int wave = tid >> 5, lane = tid & 31;
    int half = lane >> 4, r = lane & 15, ko = half * 8;

    // Q fragments stay in registers for the whole kernel (16 rows x 64 hd)
    int qrow = qb * 128 + wave * 16 + r;
    int qt   = win_tok(win, qrow);
    const __bf16* qp = qkv + (size_t)qt * (3 * DIM) + head * HD;
    Frag16 fq0, fq1;
    fq0.u[0] = *(const uint4*)(qp + ko);
    fq0.u[1] = *(const uint4*)(qp + 16 + ko);
    fq1.u[0] = *(const uint4*)(qp + 32 + ko);
    fq1.u[1] = *(const uint4*)(qp + 48 + ko);

    float mrow[8], lrow[8];
    v8f accO[4];
    for (int i = 0; i < 8; ++i) { mrow[i] = -3.0e38f; lrow[i] = 0.0f; }
    for (int t2 = 0; t2 < 4; ++t2) accO[t2] = v8f_zero();

    for (int ci = 0; ci < WTOK / 64; ++ci) {
        // stage K chunk (row-major) and V chunk (transposed)
        for (int pass = 0; pass < 2; ++pass) {
            int slot = pass * 256 + tid;
            int kr = slot >> 3, kc = (slot & 7) * 8;
            int kt = win_tok(win, ci * 64 + kr);
            const __bf16* kp = qkv + (size_t)kt * (3 * DIM) + DIM + head * HD + kc;
            *(uint4*)&Ksh[kr][kc] = *(const uint4*)kp;
            const __bf16* vp = qkv + (size_t)kt * (3 * DIM) + 2 * DIM + head * HD + kc;
            union { uint4 u; __bf16 h[8]; } vv;
            vv.u = *(const uint4*)vp;
            for (int e = 0; e < 8; ++e) Vt[kc + e][kr] = vv.h[e];
        }
        __syncthreads();

        // S = Q * K^T  (16 x 64 per wave) via 8 WMMAs
        v8f accS[4];
        for (int j = 0; j < 4; ++j) {
            accS[j] = v8f_zero();
            Frag16 fb;
            fb.u[0] = *(const uint4*)&Ksh[j * 16 + r][ko];
            fb.u[1] = *(const uint4*)&Ksh[j * 16 + r][16 + ko];
            accS[j] = wmma_bf16(fq0, fb, accS[j]);
            fb.u[0] = *(const uint4*)&Ksh[j * 16 + r][32 + ko];
            fb.u[1] = *(const uint4*)&Ksh[j * 16 + r][48 + ko];
            accS[j] = wmma_bf16(fq1, fb, accS[j]);
        }
        for (int j = 0; j < 4; ++j)
            for (int i = 0; i < 8; ++i) accS[j][i] *= 0.125f;   // hd^-0.5

        // online softmax per row (row m = i + 8*half lives in one 16-lane half)
        for (int i = 0; i < 8; ++i) {
            float mx = fmaxf(fmaxf(accS[0][i], accS[1][i]), fmaxf(accS[2][i], accS[3][i]));
            for (int off = 1; off < 16; off <<= 1) mx = fmaxf(mx, __shfl_xor(mx, off, 32));
            float mnew = fmaxf(mrow[i], mx);
            float corr = __expf(mrow[i] - mnew);
            mrow[i] = mnew;
            float rs = 0.0f;
            for (int j = 0; j < 4; ++j) {
                float p = __expf(accS[j][i] - mnew);
                accS[j][i] = p;
                rs += p;
            }
            for (int off = 1; off < 16; off <<= 1) rs += __shfl_xor(rs, off, 32);
            lrow[i] = lrow[i] * corr + rs;
            for (int t2 = 0; t2 < 4; ++t2) accO[t2][i] *= corr;
            int m = i + 8 * half;
            for (int j = 0; j < 4; ++j)
                Ps[wave][m][j * 16 + r] = (__bf16)accS[j][i];
        }
        // wave-private LDS RAW: drain DS counter before fragment reload
        asm volatile("s_wait_dscnt 0" ::: "memory");

        // O += P * V  (16 x 64 per wave) via 8 WMMAs
        for (int kt2 = 0; kt2 < 2; ++kt2) {
            Frag16 fp;
            fp.u[0] = *(const uint4*)&Ps[wave][r][kt2 * 32 + ko];
            fp.u[1] = *(const uint4*)&Ps[wave][r][kt2 * 32 + 16 + ko];
            for (int t2 = 0; t2 < 4; ++t2) {
                Frag16 fv;
                fv.u[0] = *(const uint4*)&Vt[t2 * 16 + r][kt2 * 32 + ko];
                fv.u[1] = *(const uint4*)&Vt[t2 * 16 + r][kt2 * 32 + 16 + ko];
                accO[t2] = wmma_bf16(fp, fv, accO[t2]);
            }
        }
        __syncthreads();
    }

    // normalize + scatter to (token, head*64 + hd) in global token order
    for (int i = 0; i < 8; ++i) {
        int m = i + 8 * half;
        int ot = win_tok(win, qb * 128 + wave * 16 + m);
        float inv = 1.0f / lrow[i];
        for (int t2 = 0; t2 < 4; ++t2)
            out[(size_t)ot * DIM + head * HD + t2 * 16 + r] = (__bf16)(accO[t2][i] * inv);
    }
}

// ---------------------------------------------------------------------------
// Host-side orchestration
// ---------------------------------------------------------------------------
extern "C" void kernel_launch(void* const* d_in, const int* in_sizes, int n_in,
                              void* d_out, int out_size, void* d_ws, size_t ws_size,
                              hipStream_t stream) {
    (void)in_sizes; (void)n_in; (void)out_size; (void)ws_size;
    const float* x    = (const float*)d_in[0];
    const float* g1   = (const float*)d_in[1];
    const float* b1   = (const float*)d_in[2];
    const float* aw1  = (const float*)d_in[3];
    const float* ab1  = (const float*)d_in[4];
    const float* acw  = (const float*)d_in[5];
    const float* acb  = (const float*)d_in[6];
    const float* aw2  = (const float*)d_in[7];
    const float* ab2  = (const float*)d_in[8];
    const float* qkvw = (const float*)d_in[9];
    const float* qkvb = (const float*)d_in[10];
    const float* pw   = (const float*)d_in[11];
    const float* pb   = (const float*)d_in[12];
    const float* g2   = (const float*)d_in[13];
    const float* b2   = (const float*)d_in[14];
    const float* mw1  = (const float*)d_in[15];
    const float* mb1  = (const float*)d_in[16];
    const float* mw2  = (const float*)d_in[17];
    const float* mb2  = (const float*)d_in[18];

    char* ws = (char*)d_ws;
    size_t off = 0;
    auto alloc = [&](size_t bytes) -> void* {
        void* p = (void*)(ws + off);
        off += (bytes + 255) & ~(size_t)255;
        return p;
    };
    __bf16* h      = (__bf16*)alloc((size_t)TOK * DIM * 2);   // LN1 out, then hh in place
    float*  a1     = (float*) alloc((size_t)TOK * MID * 4);
    float*  a2     = (float*) alloc((size_t)TOK * MID * 4);
    __bf16* qkv    = (__bf16*)alloc((size_t)TOK * 3 * DIM * 2);
    __bf16* attn   = (__bf16*)alloc((size_t)TOK * DIM * 2);
    float*  x2     = (float*) alloc((size_t)TOK * DIM * 4);
    __bf16* y      = (__bf16*)alloc((size_t)TOK * DIM * 2);
    __bf16* u      = (__bf16*)alloc((size_t)TOK * HID * 2);
    __bf16* qkvwT  = (__bf16*)alloc((size_t)DIM * 3 * DIM * 2);
    __bf16* pwT    = (__bf16*)alloc((size_t)DIM * DIM * 2);
    __bf16* mw1T   = (__bf16*)alloc((size_t)DIM * HID * 2);
    __bf16* mw2T   = (__bf16*)alloc((size_t)HID * DIM * 2);

    // 1) weight conversion (fp32 KxN -> bf16 NxK)
    auto cvt = [&](const float* w, __bf16* wt, int K, int N) {
        int total = K * N;
        convert_wT_kernel<<<(total + 255) / 256, 256, 0, stream>>>(w, wt, K, N);
    };
    cvt(qkvw, qkvwT, DIM, 3 * DIM);
    cvt(pw,   pwT,   DIM, DIM);
    cvt(mw1,  mw1T,  DIM, HID);
    cvt(mw2,  mw2T,  HID, DIM);

    // 2) LN1 + adapter down
    ln_kernel<true><<<TOK, 256, 0, stream>>>(x, g1, b1, h, aw1, ab1, a1);
    // 3) depthwise conv + relu
    dwconv_kernel<<<(TOK + 255) / 256, 256, 0, stream>>>(a1, acw, acb, a2);
    // 4) hh = 2h + relu(a2 @ aw2 + ab2)
    hh_kernel<<<TOK, 256, 0, stream>>>(h, a2, aw2, ab2);

    // 5) qkv GEMM: (16384x768) x (768x2304)
    gemm_bf16_kernel<0, false, true><<<dim3(3 * DIM / 128, TOK / 128), 256, 0, stream>>>(
        h, qkvwT, qkvb, nullptr, qkv, TOK, 3 * DIM, DIM);

    // 6) windowed attention
    attn_kernel<<<dim3(NWIN * NH, WTOK / 128), 256, 0, stream>>>(qkv, attn);

    // 7) proj + residual(x) -> x2 (fp32)
    gemm_bf16_kernel<0, true, false><<<dim3(DIM / 128, TOK / 128), 256, 0, stream>>>(
        attn, pwT, pb, x, x2, TOK, DIM, DIM);

    // 8) LN2
    ln_kernel<false><<<TOK, 256, 0, stream>>>(x2, g2, b2, y, nullptr, nullptr, nullptr);

    // 9) mlp1 + exact gelu -> u (bf16)
    gemm_bf16_kernel<1, false, true><<<dim3(HID / 128, TOK / 128), 256, 0, stream>>>(
        y, mw1T, mb1, nullptr, u, TOK, HID, DIM);

    // 10) mlp2 + residual(x2) -> out (fp32)
    gemm_bf16_kernel<0, true, false><<<dim3(DIM / 128, TOK / 128), 256, 0, stream>>>(
        u, mw2T, mb2, x2, d_out, TOK, DIM, HID);
}
